// InteractionGraphEncoder_4329327034975
// MI455X (gfx1250) — matrix operand order repaired and checked by reading, not compiled, not run on hardware
//
#include <hip/hip_runtime.h>
#include <hip/hip_bf16.h>

// ---------------- problem constants ----------------
#define NN   100000   // nodes
#define EE   800000   // edges per relation
#define RR   6        // relations
#define BB   1000     // graphs
#define HH   128      // hidden/out dim
#define D1P  32       // layer-1 K padded (13 cont + 8 pos -> 21 -> 32)
#define NPOSK 24
#define EPW  4        // edges per wave in scatter

typedef __attribute__((ext_vector_type(16))) __bf16 v16bf;
typedef __attribute__((ext_vector_type(8)))  __bf16 v8bf;
typedef __attribute__((ext_vector_type(4)))  __bf16 v4bf;
typedef __attribute__((ext_vector_type(8)))  float  v8f;

// order-preserving float -> uint key (for atomicMax-based segment max)
__device__ __forceinline__ unsigned fkey(float f) {
    unsigned u = __float_as_uint(f);
    return (u & 0x80000000u) ? ~u : (u | 0x80000000u);
}
__device__ __forceinline__ float fkey_dec(unsigned k) {
    unsigned u = (k & 0x80000000u) ? (k ^ 0x80000000u) : ~k;
    return __uint_as_float(u);
}
__device__ __forceinline__ void atomic_addf(float* p, float v) {
    __hip_atomic_fetch_add(p, v, __ATOMIC_RELAXED, __HIP_MEMORY_SCOPE_AGENT);
}

// ---------------- feature build: [cont(13) | pos_emb(8) | zero pad] as bf16 ----------------
__global__ void build_x1_k(const float* __restrict__ x_all,
                           const float* __restrict__ pos_table,
                           __bf16* __restrict__ Xb, int ldx) {
    int i = blockIdx.x * blockDim.x + threadIdx.x;
    if (i >= NN) return;
    const float* row = x_all + (size_t)i * 14;
    __bf16* o = Xb + (size_t)i * ldx;
    int c = 0;
    #pragma unroll
    for (int j = 0; j < 14; ++j)
        if (j != 5) o[c++] = (__bf16)row[j];
    int p = (int)row[5];
    p = p < 0 ? 0 : (p > NPOSK - 1 ? NPOSK - 1 : p);
    const float* pe = pos_table + (size_t)p * 8;
    #pragma unroll
    for (int j = 0; j < 8; ++j) o[c++] = (__bf16)pe[j];
    for (; c < ldx; ++c) o[c] = (__bf16)0.f;
}

// ---------------- weight prep: W[R][Kin][HH] f32 -> Wt[R][HH][Kp] bf16 (transposed, padded) ---
__global__ void prep_w_k(const float* __restrict__ W, int Kin, int Kp,
                         __bf16* __restrict__ Wt) {
    int idx = blockIdx.x * blockDim.x + threadIdx.x;
    int total = RR * HH * Kp;
    if (idx >= total) return;
    int k = idx % Kp;
    int n = (idx / Kp) % HH;
    int r = idx / (Kp * HH);
    Wt[idx] = (k < Kin) ? (__bf16)W[((size_t)r * Kin + k) * HH + n] : (__bf16)0.f;
}

// ---------------- WMMA GEMM: XP[N][128] = Xb[N][LDX](bf16) @ W -------------------------------
// Strip-mined: one wave computes a 16(M) x 128(N) strip; A fragment loaded once per K-step and
// reused across all 8 column tiles; B fragments software-pipelined (loads for tile tn+1 issued
// before the WMMA of tile tn). LDX is compile-time so every B-tile address is base + immediate
// offset (<= 28 KB, fits the 24-bit IOFFSET) -- no per-tile pointer registers, no spills.
template <int LDX>
__global__ void gemm_strip_k(const __bf16* __restrict__ Xb,
                             const __bf16* __restrict__ Wt,   // [HH][LDX] = W^T bf16
                             float* __restrict__ XP) {
    int gid  = blockIdx.x * blockDim.x + threadIdx.x;
    int wave = gid >> 5;
    int lane = gid & 31;
    if (wave >= NN / 16) return;
    int l = lane & 15, hi = lane >> 4;

    // A fragment (ISA 16-bit A 16x32 layout): lane holds row m=l,
    // halfs 0..7 = K (hi*8 .. hi*8+7), halfs 8..15 = K (16+hi*8 ..)
    const __bf16* arow = Xb + (size_t)(wave * 16 + l) * LDX + hi * 8;
    // B fragment base: lane holds col n=l, halfs 0..7 = K (hi*8 ..), halfs 8..15 = K (16+hi*8 ..);
    // tile tn adds the compile-time constant 16*LDX elements.
    const __bf16* wbase = Wt + (size_t)l * LDX + hi * 8;

    v8f c[8];
    #pragma unroll
    for (int t = 0; t < 8; ++t) c[t] = (v8f){};

    #pragma unroll
    for (int k0 = 0; k0 < LDX; k0 += 32) {
        v8bf alo = *(const v8bf*)(arow + k0);
        v8bf ahi = *(const v8bf*)(arow + k0 + 16);
        v16bf A;
        #pragma unroll
        for (int i = 0; i < 8; ++i) { A[i] = alo[i]; A[8 + i] = ahi[i]; }

        // prime the B pipeline with tile 0
        v8bf blo = *(const v8bf*)(wbase + k0);
        v8bf bhi = *(const v8bf*)(wbase + k0 + 16);
        #pragma unroll
        for (int tn = 0; tn < 8; ++tn) {
            v8bf nlo, nhi;
            if (tn < 7) {                       // issue next tile's loads first
                const __bf16* nb = wbase + (tn + 1) * 16 * LDX + k0;
                nlo = *(const v8bf*)(nb);
                nhi = *(const v8bf*)(nb + 16);
            }
            v16bf B;
            #pragma unroll
            for (int i = 0; i < 8; ++i) { B[i] = blo[i]; B[8 + i] = bhi[i]; }
            c[tn] = __builtin_amdgcn_wmma_f32_16x16x32_bf16(false, A, false, B,
                                                            (short)0, c[tn], false, false);
            if (tn < 7) { blo = nlo; bhi = nhi; }
        }
    }
    // D layout: VGPR g, lanes 0-15 -> M=g, lanes 16-31 -> M=8+g; N = l
    #pragma unroll
    for (int tn = 0; tn < 8; ++tn) {
        float* out = XP + (size_t)(wave * 16 + hi * 8) * HH + tn * 16 + l;
        #pragma unroll
        for (int g = 0; g < 8; ++g) out[(size_t)g * HH] = c[tn][g];
    }
}

// ---------------- per-node dual dot products (wave per node) ----------------
__global__ void node_scores_k(const float* __restrict__ XP,
                              const float* __restrict__ a_s,
                              const float* __restrict__ a_d,
                              float* __restrict__ zs, float* __restrict__ zd) {
    int gid = blockIdx.x * blockDim.x + threadIdx.x;
    int node = gid >> 5, lane = gid & 31;
    if (node >= NN) return;
    float4 v = ((const float4*)(XP + (size_t)node * HH))[lane];
    float4 s = ((const float4*)a_s)[lane];
    float4 d = ((const float4*)a_d)[lane];
    float ps = v.x * s.x + v.y * s.y + v.z * s.z + v.w * s.w;
    float pd = v.x * d.x + v.y * d.y + v.z * d.z + v.w * d.w;
    #pragma unroll
    for (int o = 16; o; o >>= 1) { ps += __shfl_xor(ps, o); pd += __shfl_xor(pd, o); }
    if (lane == 0) { zs[node] = ps; zd[node] = pd; }
}

__global__ void zero_k(unsigned* __restrict__ p, int n) {
    int i = blockIdx.x * blockDim.x + threadIdx.x;
    if (i < n) p[i] = 0u;
}

__global__ void bsum_k(const float* __restrict__ b, float* __restrict__ bsum) {
    int j = threadIdx.x;
    float s = 0.f;
    #pragma unroll
    for (int r = 0; r < RR; ++r) s += b[r * HH + j];
    bsum[j] = s;
}

__global__ void init_acc_k(float* __restrict__ acc, const float* __restrict__ bsum) {
    int i = blockIdx.x * blockDim.x + threadIdx.x;
    if (i < NN * HH) acc[i] = bsum[i & (HH - 1)];
}

// ---------------- edge passes: LeakyReLU logit + segment max / exp-sum / scatter ------------
__global__ void edge_max_k(const int* __restrict__ src, const int* __restrict__ dst,
                           const float* __restrict__ zs, const float* __restrict__ zd,
                           float* __restrict__ zb, unsigned* __restrict__ smax) {
    int e = blockIdx.x * blockDim.x + threadIdx.x;
    if (e >= EE) return;
    int d = dst[e];
    float z = zs[src[e]] + zd[d];
    z = z > 0.f ? z : 0.2f * z;          // LeakyReLU(0.2)
    zb[e] = z;
    atomicMax(smax + d, fkey(z));
}

__global__ void edge_exp_k(const int* __restrict__ dst,
                           float* __restrict__ zb,        // in: z, out: exp(z-max)
                           const unsigned* __restrict__ smax,
                           float* __restrict__ ssum) {
    int e = blockIdx.x * blockDim.x + threadIdx.x;
    if (e >= EE) return;
    int d = dst[e];
    float ex = __expf(zb[e] - fkey_dec(smax[d]));
    zb[e] = ex;
    atomic_addf(ssum + d, ex);
}

// wave processes EPW consecutive edges; prefetch next src row while scattering current
__global__ void edge_scatter_k(const int* __restrict__ src, const int* __restrict__ dst,
                               const float* __restrict__ eb, const float* __restrict__ ssum,
                               const float* __restrict__ XP, float* __restrict__ acc) {
    int gid = blockIdx.x * blockDim.x + threadIdx.x;
    int wave = gid >> 5, lane = gid & 31;
    int e0 = wave * EPW;
    if (e0 >= EE) return;
    #pragma unroll
    for (int i = 0; i < EPW; ++i) {
        int e = e0 + i;
        if (e >= EE) break;
        if (i + 1 < EPW && e + 1 < EE)   // hide next gather behind current atomics
            __builtin_prefetch(XP + (size_t)src[e + 1] * HH + lane * 4, 0, 3);
        int s = src[e], d = dst[e];
        float alpha = eb[e] / ssum[d];
        float4 v = ((const float4*)(XP + (size_t)s * HH))[lane];
        float* o = acc + (size_t)d * HH + lane * 4;
        atomic_addf(o + 0, alpha * v.x);
        atomic_addf(o + 1, alpha * v.y);
        atomic_addf(o + 2, alpha * v.z);
        atomic_addf(o + 3, alpha * v.w);
    }
}

// ---------------- exact GELU + LayerNorm (wave per node) ----------------
__global__ void act_ln_k(const float* __restrict__ acc,
                         const float* __restrict__ lw, const float* __restrict__ lb,
                         __bf16* __restrict__ xb, int ldx, int wr_bf16,
                         float* __restrict__ hout, int wr_f32) {
    int gid = blockIdx.x * blockDim.x + threadIdx.x;
    int node = gid >> 5, lane = gid & 31;
    if (node >= NN) return;
    float4 v = ((const float4*)(acc + (size_t)node * HH))[lane];
    float g0 = 0.5f * v.x * (1.f + erff(v.x * 0.70710678118f));
    float g1 = 0.5f * v.y * (1.f + erff(v.y * 0.70710678118f));
    float g2 = 0.5f * v.z * (1.f + erff(v.z * 0.70710678118f));
    float g3 = 0.5f * v.w * (1.f + erff(v.w * 0.70710678118f));
    float s = g0 + g1 + g2 + g3;
    #pragma unroll
    for (int o = 16; o; o >>= 1) s += __shfl_xor(s, o);
    float mean = s * (1.f / HH);
    float c0 = g0 - mean, c1 = g1 - mean, c2 = g2 - mean, c3 = g3 - mean;
    float q = c0 * c0 + c1 * c1 + c2 * c2 + c3 * c3;
    #pragma unroll
    for (int o = 16; o; o >>= 1) q += __shfl_xor(q, o);
    float inv = rsqrtf(q * (1.f / HH) + 1e-5f);
    int j = lane * 4;
    float y0 = c0 * inv * lw[j + 0] + lb[j + 0];
    float y1 = c1 * inv * lw[j + 1] + lb[j + 1];
    float y2 = c2 * inv * lw[j + 2] + lb[j + 2];
    float y3 = c3 * inv * lw[j + 3] + lb[j + 3];
    if (wr_bf16) {
        v4bf o4; o4[0] = (__bf16)y0; o4[1] = (__bf16)y1; o4[2] = (__bf16)y2; o4[3] = (__bf16)y3;
        *(v4bf*)(xb + (size_t)node * ldx + j) = o4;
    }
    if (wr_f32)
        ((float4*)(hout + (size_t)node * HH))[lane] = make_float4(y0, y1, y2, y3);
}

// ---------------- attention pooling over batch segments ----------------
__global__ void pool_max_k(const int* __restrict__ batch, const float* __restrict__ sc,
                           unsigned* __restrict__ gmax) {
    int i = blockIdx.x * blockDim.x + threadIdx.x;
    if (i >= NN) return;
    atomicMax(gmax + batch[i], fkey(sc[i]));
}
__global__ void pool_exp_k(const int* __restrict__ batch, const float* __restrict__ sc,
                           const unsigned* __restrict__ gmax,
                           float* __restrict__ ew, float* __restrict__ gsum) {
    int i = blockIdx.x * blockDim.x + threadIdx.x;
    if (i >= NN) return;
    int b = batch[i];
    float ex = __expf(sc[i] - fkey_dec(gmax[b]));
    ew[i] = ex;
    atomic_addf(gsum + b, ex);
}
__global__ void pool_scatter_k(const int* __restrict__ batch, const float* __restrict__ ew,
                               const float* __restrict__ gsum, const float* __restrict__ h,
                               float* __restrict__ pool) {
    int gid = blockIdx.x * blockDim.x + threadIdx.x;
    int node = gid >> 5, lane = gid & 31;
    if (node >= NN) return;
    int b = batch[node];
    float w = ew[node] / gsum[b];
    float4 v = ((const float4*)(h + (size_t)node * HH))[lane];
    float* o = pool + (size_t)b * HH + lane * 4;
    atomic_addf(o + 0, w * v.x);
    atomic_addf(o + 1, w * v.y);
    atomic_addf(o + 2, w * v.z);
    atomic_addf(o + 3, w * v.w);
}

// ---------------- final projection: out[B][128] = pool @ projW + projb ----------------
__global__ void proj_k(const float* __restrict__ pool, const float* __restrict__ projW,
                       const float* __restrict__ projb, float* __restrict__ out) {
    __shared__ float sp[HH];
    int b = blockIdx.x, t = threadIdx.x;
    sp[t] = pool[(size_t)b * HH + t];
    __syncthreads();
    float a = projb[t];
    #pragma unroll 8
    for (int k = 0; k < HH; ++k) a += sp[k] * projW[(size_t)k * HH + t];
    out[(size_t)b * HH + t] = a;
}

// =====================================================================================
extern "C" void kernel_launch(void* const* d_in, const int* in_sizes, int n_in,
                              void* d_out, int out_size, void* d_ws, size_t ws_size,
                              hipStream_t stream) {
    const float* x_all     = (const float*)d_in[0];
    const int*   edges     = (const int*)d_in[1];   // [R][2][E]
    const int*   batch     = (const int*)d_in[2];
    const float* pos_table = (const float*)d_in[3];
    const float* W1  = (const float*)d_in[4];
    const float* as1 = (const float*)d_in[5];
    const float* ad1 = (const float*)d_in[6];
    const float* b1  = (const float*)d_in[7];
    const float* W2  = (const float*)d_in[8];
    const float* as2 = (const float*)d_in[9];
    const float* ad2 = (const float*)d_in[10];
    const float* b2  = (const float*)d_in[11];
    const float* n1w = (const float*)d_in[12];
    const float* n1b = (const float*)d_in[13];
    const float* n2w = (const float*)d_in[14];
    const float* n2b = (const float*)d_in[15];
    const float* query = (const float*)d_in[16];
    const float* projW = (const float*)d_in[17];
    const float* projb = (const float*)d_in[18];
    float* out = (float*)d_out;
    (void)in_sizes; (void)n_in; (void)out_size; (void)ws_size;

    // ---- workspace carve-out (~134 MB total) ----
    char* ws = (char*)d_ws;
    size_t off = 0;
    auto take = [&](size_t bytes) -> char* {
        char* p = ws + off;
        off = (off + bytes + 255) & ~(size_t)255;
        return p;
    };
    __bf16*   Xb   = (__bf16*)take((size_t)NN * HH * 2);     // bf16 activations (WMMA A)
    __bf16*   Wt   = (__bf16*)take((size_t)RR * HH * HH * 2);// bf16 W^T (WMMA B)
    float*    XP   = (float*)take((size_t)NN * HH * 4);      // projected feats / final h
    float*    ACC  = (float*)take((size_t)NN * HH * 4);      // GAT accumulator
    float*    ZS   = (float*)take((size_t)NN * 4);
    float*    ZD   = (float*)take((size_t)NN * 4);
    unsigned* SMAX = (unsigned*)take((size_t)NN * 4);
    float*    SSUM = (float*)take((size_t)NN * 4);
    float*    EB   = (float*)take((size_t)EE * 4);
    float*    BS   = (float*)take(HH * 4);
    unsigned* GMAX = (unsigned*)take(BB * 4);
    float*    GSUM = (float*)take(BB * 4);
    float*    POOL = (float*)take((size_t)BB * HH * 4);

    const dim3 blk(256);
    const int gem_blocks   = ((NN / 16) * 32 + 255) / 256;              // 782 (strip waves)
    const int wave_node_bl = (NN * 32 + 255) / 256;                     // 12500
    const int scat_blocks  = (((EE + EPW - 1) / EPW) * 32 + 255) / 256; // 25000
    const int node_bl      = (NN + 255) / 256;
    const int edge_bl      = (EE + 255) / 256;

    // ================= layer 1 =================
    build_x1_k<<<node_bl, blk, 0, stream>>>(x_all, pos_table, Xb, D1P);
    prep_w_k<<<(RR * HH * D1P + 255) / 256, blk, 0, stream>>>(W1, 21, D1P, Wt);
    bsum_k<<<1, HH, 0, stream>>>(b1, BS);
    init_acc_k<<<(NN * HH + 255) / 256, blk, 0, stream>>>(ACC, BS);
    for (int r = 0; r < RR; ++r) {
        gemm_strip_k<D1P><<<gem_blocks, blk, 0, stream>>>(Xb, Wt + (size_t)r * HH * D1P, XP);
        node_scores_k<<<wave_node_bl, blk, 0, stream>>>(XP, as1 + r * HH, ad1 + r * HH, ZS, ZD);
        zero_k<<<node_bl, blk, 0, stream>>>(SMAX, NN);
        zero_k<<<node_bl, blk, 0, stream>>>((unsigned*)SSUM, NN);
        const int* src = edges + (size_t)r * 2 * EE;
        const int* dst = src + EE;
        edge_max_k<<<edge_bl, blk, 0, stream>>>(src, dst, ZS, ZD, EB, SMAX);
        edge_exp_k<<<edge_bl, blk, 0, stream>>>(dst, EB, SMAX, SSUM);
        edge_scatter_k<<<scat_blocks, blk, 0, stream>>>(src, dst, EB, SSUM, XP, ACC);
    }
    act_ln_k<<<wave_node_bl, blk, 0, stream>>>(ACC, n1w, n1b, Xb, HH, 1, XP, 0);

    // ================= layer 2 =================
    prep_w_k<<<(RR * HH * HH + 255) / 256, blk, 0, stream>>>(W2, HH, HH, Wt);
    bsum_k<<<1, HH, 0, stream>>>(b2, BS);
    init_acc_k<<<(NN * HH + 255) / 256, blk, 0, stream>>>(ACC, BS);
    for (int r = 0; r < RR; ++r) {
        gemm_strip_k<HH><<<gem_blocks, blk, 0, stream>>>(Xb, Wt + (size_t)r * HH * HH, XP);
        node_scores_k<<<wave_node_bl, blk, 0, stream>>>(XP, as2 + r * HH, ad2 + r * HH, ZS, ZD);
        zero_k<<<node_bl, blk, 0, stream>>>(SMAX, NN);
        zero_k<<<node_bl, blk, 0, stream>>>((unsigned*)SSUM, NN);
        const int* src = edges + (size_t)r * 2 * EE;
        const int* dst = src + EE;
        edge_max_k<<<edge_bl, blk, 0, stream>>>(src, dst, ZS, ZD, EB, SMAX);
        edge_exp_k<<<edge_bl, blk, 0, stream>>>(dst, EB, SMAX, SSUM);
        edge_scatter_k<<<scat_blocks, blk, 0, stream>>>(src, dst, EB, SSUM, XP, ACC);
    }
    act_ln_k<<<wave_node_bl, blk, 0, stream>>>(ACC, n2w, n2b, Xb, HH, 0, XP, 1); // XP = final h f32

    // ================= attention pooling + projection =================
    node_scores_k<<<wave_node_bl, blk, 0, stream>>>(XP, query, query, ZS, ZD); // ZS = scores
    zero_k<<<(BB + 255) / 256, blk, 0, stream>>>(GMAX, BB);
    zero_k<<<(BB + 255) / 256, blk, 0, stream>>>((unsigned*)GSUM, BB);
    zero_k<<<(BB * HH + 255) / 256, blk, 0, stream>>>((unsigned*)POOL, BB * HH);
    pool_max_k<<<node_bl, blk, 0, stream>>>(batch, ZS, GMAX);
    pool_exp_k<<<node_bl, blk, 0, stream>>>(batch, ZS, GMAX, ZD, GSUM);
    pool_scatter_k<<<wave_node_bl, blk, 0, stream>>>(batch, ZD, GSUM, XP, POOL);
    proj_k<<<BB, HH, 0, stream>>>(POOL, projW, projb, out);
}
